// OperationBranch_35562329211642
// MI455X (gfx1250) — matrix operand order repaired and checked by reading, not compiled, not run on hardware
//
#include <hip/hip_runtime.h>
#include <hip/hip_bf16.h>
#include <math.h>

typedef __attribute__((ext_vector_type(16))) _Float16 v16h;
typedef __attribute__((ext_vector_type(8)))  _Float16 v8h;
typedef __attribute__((ext_vector_type(8)))  float    v8f;

#define BN      4
#define SEQ     2048
#define DIM     512
#define NH      8
#define HDK     64
#define FFN_DIM 2048
#define MROWS   (BN * SEQ)   // 8192

// ---------------------------------------------------------------------------
// WMMA fragment helpers (wave32, layouts per CDNA5 ISA 7.12.2)
// ---------------------------------------------------------------------------
__device__ __forceinline__ v16h pack16(v8h lo, v8h hi) {
  v16h r;
#pragma unroll
  for (int i = 0; i < 8; ++i) { r[i] = lo[i]; r[i + 8] = hi[i]; }
  return r;
}

// A fragment (16x32 f16): rows row0..row0+15 of row-major A (stride lda),
// reduction k0..k0+31.  Per-lane: two contiguous 16B loads.
__device__ __forceinline__ v16h load_a_frag(const _Float16* __restrict__ A,
                                            int lda, int row0, int k0) {
  int lane = threadIdx.x & 31;
  int m  = lane & 15;
  int kb = k0 + ((lane & 16) ? 8 : 0);
  const _Float16* p = A + (size_t)(row0 + m) * lda + kb;
  v8h lo = *(const v8h*)p;          // K = kb .. kb+7
  v8h hi = *(const v8h*)(p + 16);   // K = kb+16 .. kb+23
  return pack16(lo, hi);
}

// B fragment (32x16 f16) built from the TRANSPOSED operand BT (row-major
// [Nout, K], stride ldb): output cols n0..n0+15, reduction k0..k0+31.
__device__ __forceinline__ v16h load_b_frag(const _Float16* __restrict__ BT,
                                            int ldb, int n0, int k0) {
  int lane = threadIdx.x & 31;
  int n  = lane & 15;
  int kb = k0 + ((lane & 16) ? 8 : 0);
  const _Float16* p = BT + (size_t)(n0 + n) * ldb + kb;
  v8h lo = *(const v8h*)p;
  v8h hi = *(const v8h*)(p + 16);
  return pack16(lo, hi);
}

__device__ __forceinline__ v8f wmma16(v16h a, v16h b, v8f c) {
  return __builtin_amdgcn_wmma_f32_16x16x32_f16(false, a, false, b,
                                                (short)0, c, false, false);
}

__device__ __forceinline__ v8f vzero8() {
  v8f z = {0.f, 0.f, 0.f, 0.f, 0.f, 0.f, 0.f, 0.f};
  return z;
}

// ---------------------------------------------------------------------------
// Weight transpose + f16 downconvert:  WT[n][k] = (f16) W[k][n]
// ---------------------------------------------------------------------------
__global__ void transpose_to_f16(const float* __restrict__ W,
                                 _Float16* __restrict__ WT, int K, int Nout) {
  int idx = blockIdx.x * blockDim.x + threadIdx.x;
  if (idx >= K * Nout) return;
  int n = idx / K;
  int k = idx - n * K;
  WT[idx] = (_Float16)W[(size_t)k * Nout + n];
}

// ---------------------------------------------------------------------------
// LayerNorm over rows of DIM, fp32 in -> f16 out.  One block (256 thr) / row.
// ---------------------------------------------------------------------------
__global__ void layernorm_f16(const float* __restrict__ X,
                              const float* __restrict__ g,
                              const float* __restrict__ beta,
                              _Float16* __restrict__ out) {
  __shared__ float red[256];
  int row = blockIdx.x, tid = threadIdx.x;
  const float* xr = X + (size_t)row * DIM;
  float v0 = xr[tid], v1 = xr[tid + 256];
  red[tid] = v0 + v1;
  __syncthreads();
  for (int off = 128; off > 0; off >>= 1) {
    if (tid < off) red[tid] += red[tid + off];
    __syncthreads();
  }
  float mu = red[0] * (1.0f / DIM);
  __syncthreads();
  float d0 = v0 - mu, d1 = v1 - mu;
  red[tid] = d0 * d0 + d1 * d1;
  __syncthreads();
  for (int off = 128; off > 0; off >>= 1) {
    if (tid < off) red[tid] += red[tid + off];
    __syncthreads();
  }
  float rstd = rsqrtf(red[0] * (1.0f / DIM) + 1e-5f);
  _Float16* orow = out + (size_t)row * DIM;
  orow[tid]       = (_Float16)(d0 * rstd * g[tid] + beta[tid]);
  orow[tid + 256] = (_Float16)(d1 * rstd * g[tid + 256] + beta[tid + 256]);
}

// ---------------------------------------------------------------------------
// Generic WMMA GEMM:  out = act(A @ B + bias) [+ resid]
// A: f16 [M,K] row-major; BT: f16 [Nout,K] row-major (pre-transposed B).
// Block = 128 threads (4 waves); each wave owns a 16x64 output tile.
// ---------------------------------------------------------------------------
template <bool RELU, bool RESID, bool OUTF16>
__global__ void gemm_wmma(const _Float16* __restrict__ A,
                          const _Float16* __restrict__ BT,
                          const float* __restrict__ bias,
                          const float* __restrict__ resid,
                          _Float16* __restrict__ outH,
                          float* __restrict__ outF, int K, int Nout) {
  int wave = threadIdx.x >> 5;
  int lane = threadIdx.x & 31;
  int r0 = blockIdx.y * 64 + wave * 16;
  int c0 = blockIdx.x * 64;

  v8f acc[4];
#pragma unroll
  for (int t = 0; t < 4; ++t) acc[t] = vzero8();

  for (int k0 = 0; k0 < K; k0 += 32) {
    v16h a = load_a_frag(A, K, r0, k0);
    if (k0 + 32 < K)  // next-chunk prefetch (global_prefetch_b8 path)
      __builtin_prefetch((const void*)(A + (size_t)(r0 + (lane & 15)) * K + k0 + 32), 0, 1);
#pragma unroll
    for (int t = 0; t < 4; ++t) {
      v16h b = load_b_frag(BT, K, c0 + 16 * t, k0);
      acc[t] = wmma16(a, b, acc[t]);
    }
  }

  int hi8 = (lane & 16) ? 8 : 0;
  int n = lane & 15;
#pragma unroll
  for (int t = 0; t < 4; ++t) {
    int col = c0 + 16 * t + n;
    float bcol = bias[col];
#pragma unroll
    for (int r = 0; r < 8; ++r) {
      int row = r0 + r + hi8;
      float v = acc[t][r] + bcol;
      if (RELU) v = fmaxf(v, 0.0f);
      if (RESID) v += resid[(size_t)row * Nout + col];
      if (OUTF16) outH[(size_t)row * Nout + col] = (_Float16)v;
      else        outF[(size_t)row * Nout + col] = v;
    }
  }
}

// ---------------------------------------------------------------------------
// Fused QKV projection + bias + RoPE.
// blockIdx.z selects Q/K/V.  Each wave: 16 rows x one head (64 cols).
// Q,K written [B,H,N,64] f16 (RoPE applied); V written transposed [B,H,64,N].
// ---------------------------------------------------------------------------
__global__ void qkv_rope_kernel(const _Float16* __restrict__ XN,
                                const _Float16* __restrict__ wqT,
                                const _Float16* __restrict__ wkT,
                                const _Float16* __restrict__ wvT,
                                const float* __restrict__ bq,
                                const float* __restrict__ bk,
                                const float* __restrict__ bv,
                                const int* __restrict__ positions,
                                _Float16* __restrict__ Qo,
                                _Float16* __restrict__ Ko,
                                _Float16* __restrict__ VTo) {
  int which = blockIdx.z;  // 0=Q, 1=K, 2=V
  const _Float16* WT = (which == 0) ? wqT : (which == 1) ? wkT : wvT;
  const float* bias  = (which == 0) ? bq  : (which == 1) ? bk  : bv;

  int wave = threadIdx.x >> 5;
  int lane = threadIdx.x & 31;
  int r0 = blockIdx.y * 64 + wave * 16;
  int h  = blockIdx.x;
  int c0 = h * HDK;

  v8f acc[4];
#pragma unroll
  for (int t = 0; t < 4; ++t) acc[t] = vzero8();

  for (int k0 = 0; k0 < DIM; k0 += 32) {
    v16h a = load_a_frag(XN, DIM, r0, k0);
#pragma unroll
    for (int t = 0; t < 4; ++t) {
      v16h b = load_b_frag(WT, DIM, c0 + 16 * t, k0);
      acc[t] = wmma16(a, b, acc[t]);
    }
  }

  int hi8 = (lane & 16) ? 8 : 0;
  int n = lane & 15;
  int b = r0 / SEQ;
  int nq0 = r0 % SEQ;

  float y[4][8];
#pragma unroll
  for (int t = 0; t < 4; ++t) {
    float bcol = bias[c0 + 16 * t + n];
#pragma unroll
    for (int r = 0; r < 8; ++r) y[t][r] = acc[t][r] + bcol;
  }

  if (which == 2) {
    // V: store transposed [B,H,DK,N]; per lane 8 consecutive n-positions.
#pragma unroll
    for (int t = 0; t < 4; ++t) {
      int dk = 16 * t + n;
      v8h pk;
#pragma unroll
      for (int r = 0; r < 8; ++r) pk[r] = (_Float16)y[t][r];
      size_t base = ((size_t)(b * NH + h) * HDK + dk) * SEQ + nq0 + hi8;
      *(v8h*)(VTo + base) = pk;
    }
  } else {
    _Float16* O = (which == 0) ? Qo : Ko;
    // inv_freq for i = dk % 32; tiles 0,2 -> i=n ; tiles 1,3 -> i=16+n
    const float LN1E4_OVER = 9.210340372f / 64.0f;  // ln(10000)/64
    float invfa = __expf(-(2.0f * (float)n)        * LN1E4_OVER);
    float invfb = __expf(-(2.0f * (float)(n + 16)) * LN1E4_OVER);
#pragma unroll
    for (int r = 0; r < 8; ++r) {
      int nq = nq0 + r + hi8;
      float pos = (float)positions[b * SEQ + nq];
      float aa = pos * invfa, ab = pos * invfb;
      float sa = __sinf(aa), ca = __cosf(aa);
      float sb = __sinf(ab), cb = __cosf(ab);
      float o0 = y[0][r] * ca - y[2][r] * sa;   // dk <  32: rot = -x[dk+32]
      float o1 = y[1][r] * cb - y[3][r] * sb;
      float o2 = y[2][r] * ca + y[0][r] * sa;   // dk >= 32: rot = +x[dk-32]
      float o3 = y[3][r] * cb + y[1][r] * sb;
      size_t rowbase = ((size_t)((b * NH + h) * SEQ) + nq) * HDK;
      O[rowbase + n]      = (_Float16)o0;
      O[rowbase + 16 + n] = (_Float16)o1;
      O[rowbase + 32 + n] = (_Float16)o2;
      O[rowbase + 48 + n] = (_Float16)o3;
    }
  }
}

// ---------------------------------------------------------------------------
// Flash attention, wave32 WMMA.  Wave = 16 queries x DK=64, 32 keys / iter.
// P (probabilities) is transposed C-layout -> A-layout via per-wave LDS tile.
// Output written as f16 [B,N, H*DK] ready for the Wo GEMM.
// ---------------------------------------------------------------------------
__global__ void attention_kernel(const _Float16* __restrict__ Q,
                                 const _Float16* __restrict__ Kc,
                                 const _Float16* __restrict__ VT,
                                 const unsigned char* __restrict__ mask,
                                 _Float16* __restrict__ attnh) {
  __shared__ __align__(16) _Float16 Pbuf[4][16 * 32];

  int wave = threadIdx.x >> 5;
  int lane = threadIdx.x & 31;
  int b = blockIdx.z, h = blockIdx.y;
  int q0 = (blockIdx.x * 4 + wave) * 16;

  const _Float16* Qh = Q  + (size_t)(b * NH + h) * SEQ * HDK;
  const _Float16* Kh = Kc + (size_t)(b * NH + h) * SEQ * HDK;
  const _Float16* Vh = VT + (size_t)(b * NH + h) * HDK * SEQ;
  const unsigned char* Mb = mask + (size_t)b * SEQ * SEQ;

  v16h aq0 = load_a_frag(Qh, HDK, q0, 0);
  v16h aq1 = load_a_frag(Qh, HDK, q0, 32);

  v8f o[4];
#pragma unroll
  for (int t = 0; t < 4; ++t) o[t] = vzero8();

  const float INF = __builtin_inff();
  const float SCALE = 0.125f;       // 1/sqrt(64)
  const float LOG2E = 1.44269504f;
  float rowmax[8], rowsum[8];
#pragma unroll
  for (int r = 0; r < 8; ++r) { rowmax[r] = -INF; rowsum[r] = 0.0f; }

  int hi8 = (lane & 16) ? 8 : 0;
  int n = lane & 15;

  for (int kt = 0; kt < SEQ; kt += 32) {
    // --- scores: two 16x16 tiles, reduction over DK=64 (2 WMMA each) -------
    float s[2][8];
#pragma unroll
    for (int tt = 0; tt < 2; ++tt) {
      v8f c = vzero8();
      c = wmma16(aq0, load_b_frag(Kh, HDK, kt + 16 * tt, 0), c);
      c = wmma16(aq1, load_b_frag(Kh, HDK, kt + 16 * tt, 32), c);
      int key = kt + 16 * tt + n;
#pragma unroll
      for (int r = 0; r < 8; ++r) {
        int qrow = q0 + r + hi8;
        bool ok = Mb[(size_t)qrow * SEQ + key] != 0;
        s[tt][r] = ok ? c[r] * SCALE : -INF;
      }
    }
    // --- online softmax row stats (reduce across 16 lanes of half-wave) ----
    float csc[8];
#pragma unroll
    for (int r = 0; r < 8; ++r) {
      float mx = fmaxf(s[0][r], s[1][r]);
#pragma unroll
      for (int d = 8; d >= 1; d >>= 1) mx = fmaxf(mx, __shfl_xor(mx, d, 32));
      float mnew = fmaxf(rowmax[r], mx);
      csc[r] = (rowmax[r] == -INF) ? 0.0f : exp2f((rowmax[r] - mnew) * LOG2E);
      float p0 = (s[0][r] == -INF) ? 0.0f : exp2f((s[0][r] - mnew) * LOG2E);
      float p1 = (s[1][r] == -INF) ? 0.0f : exp2f((s[1][r] - mnew) * LOG2E);
      s[0][r] = p0; s[1][r] = p1;
      float ps = p0 + p1;
#pragma unroll
      for (int d = 8; d >= 1; d >>= 1) ps += __shfl_xor(ps, d, 32);
      rowsum[r] = rowsum[r] * csc[r] + ps;
      rowmax[r] = mnew;
    }
#pragma unroll
    for (int t = 0; t < 4; ++t)
#pragma unroll
      for (int r = 0; r < 8; ++r) o[t][r] *= csc[r];

    // --- transpose P (C-layout) into A-fragment layout via LDS -------------
#pragma unroll
    for (int tt = 0; tt < 2; ++tt)
#pragma unroll
      for (int r = 0; r < 8; ++r)
        Pbuf[wave][(r + hi8) * 32 + 16 * tt + n] = (_Float16)s[tt][r];
    // same-wave DS ordering; no cross-wave sharing -> no barrier needed
    {
      int m = lane & 15;
      int kb = (lane & 16) ? 8 : 0;
      v8h lo = *(v8h*)&Pbuf[wave][m * 32 + kb];
      v8h hi = *(v8h*)&Pbuf[wave][m * 32 + kb + 16];
      v16h ap = pack16(lo, hi);
      // --- O += P @ V  (V pre-transposed: contiguous over keys) ------------
#pragma unroll
      for (int t = 0; t < 4; ++t)
        o[t] = wmma16(ap, load_b_frag(Vh, SEQ, 16 * t, kt), o[t]);
    }
  }

  // --- finalize: divide by row sums, NaN(fully masked) -> 0, store f16 -----
#pragma unroll
  for (int t = 0; t < 4; ++t) {
    int dk = 16 * t + n;
#pragma unroll
    for (int r = 0; r < 8; ++r) {
      int qrow = q0 + r + hi8;
      float denom = rowsum[r];
      float v = (denom > 0.0f) ? (o[t][r] / denom) : 0.0f;
      attnh[((size_t)(b * SEQ + qrow)) * DIM + h * HDK + dk] = (_Float16)v;
    }
  }
}

// ---------------------------------------------------------------------------
// Host-side orchestration
// ---------------------------------------------------------------------------
#define OFF_XN    ((size_t)0)
#define OFF_WQT   (OFF_XN    + (size_t)MROWS * DIM * 2)
#define OFF_WKT   (OFF_WQT   + (size_t)DIM * DIM * 2)
#define OFF_WVT   (OFF_WKT   + (size_t)DIM * DIM * 2)
#define OFF_WOT   (OFF_WVT   + (size_t)DIM * DIM * 2)
#define OFF_W1T   (OFF_WOT   + (size_t)DIM * DIM * 2)
#define OFF_W2T   (OFF_W1T   + (size_t)DIM * FFN_DIM * 2)
#define OFF_Q     (OFF_W2T   + (size_t)FFN_DIM * DIM * 2)
#define OFF_K     (OFF_Q     + (size_t)MROWS * HDK * NH * 2 / NH * NH)  // 8 MB
#define OFF_VT    (OFF_K     + (size_t)BN * NH * SEQ * HDK * 2)
#define OFF_ATT   (OFF_VT    + (size_t)BN * NH * SEQ * HDK * 2)
#define OFF_X2    (OFF_ATT   + (size_t)MROWS * DIM * 2)
#define OFF_H2    (OFF_X2    + (size_t)MROWS * DIM * 4)
#define OFF_FFN1  (OFF_H2    + (size_t)MROWS * DIM * 2)

extern "C" void kernel_launch(void* const* d_in, const int* in_sizes, int n_in,
                              void* d_out, int out_size, void* d_ws,
                              size_t ws_size, hipStream_t stream) {
  (void)in_sizes; (void)n_in; (void)out_size; (void)ws_size;
  const float*         x         = (const float*)d_in[0];
  const unsigned char* mask      = (const unsigned char*)d_in[1];
  const int*           positions = (const int*)d_in[2];
  const float* wq = (const float*)d_in[3];  const float* bq = (const float*)d_in[4];
  const float* wk = (const float*)d_in[5];  const float* bk = (const float*)d_in[6];
  const float* wv = (const float*)d_in[7];  const float* bv = (const float*)d_in[8];
  const float* wo = (const float*)d_in[9];  const float* bo = (const float*)d_in[10];
  const float* ln1g = (const float*)d_in[11]; const float* ln1b = (const float*)d_in[12];
  const float* ln2g = (const float*)d_in[13]; const float* ln2b = (const float*)d_in[14];
  const float* w1 = (const float*)d_in[15]; const float* b1 = (const float*)d_in[16];
  const float* w2 = (const float*)d_in[17]; const float* b2 = (const float*)d_in[18];

  char* ws = (char*)d_ws;
  _Float16* xn    = (_Float16*)(ws + OFF_XN);
  _Float16* wqT   = (_Float16*)(ws + OFF_WQT);
  _Float16* wkT   = (_Float16*)(ws + OFF_WKT);
  _Float16* wvT   = (_Float16*)(ws + OFF_WVT);
  _Float16* woT   = (_Float16*)(ws + OFF_WOT);
  _Float16* w1T   = (_Float16*)(ws + OFF_W1T);
  _Float16* w2T   = (_Float16*)(ws + OFF_W2T);
  _Float16* Qb    = (_Float16*)(ws + OFF_Q);
  _Float16* Kb    = (_Float16*)(ws + OFF_K);
  _Float16* VTb   = (_Float16*)(ws + OFF_VT);
  _Float16* attnh = (_Float16*)(ws + OFF_ATT);
  float*    x2    = (float*)   (ws + OFF_X2);
  _Float16* h2    = (_Float16*)(ws + OFF_H2);
  _Float16* ffn1  = (_Float16*)(ws + OFF_FFN1);
  float*    outF  = (float*)d_out;

  // 1) weight transposes (f32 -> f16, [K,N] -> [N,K])
  transpose_to_f16<<<(DIM * DIM + 255) / 256, 256, 0, stream>>>(wq, wqT, DIM, DIM);
  transpose_to_f16<<<(DIM * DIM + 255) / 256, 256, 0, stream>>>(wk, wkT, DIM, DIM);
  transpose_to_f16<<<(DIM * DIM + 255) / 256, 256, 0, stream>>>(wv, wvT, DIM, DIM);
  transpose_to_f16<<<(DIM * DIM + 255) / 256, 256, 0, stream>>>(wo, woT, DIM, DIM);
  transpose_to_f16<<<(DIM * FFN_DIM + 255) / 256, 256, 0, stream>>>(w1, w1T, DIM, FFN_DIM);
  transpose_to_f16<<<(FFN_DIM * DIM + 255) / 256, 256, 0, stream>>>(w2, w2T, FFN_DIM, DIM);

  // 2) LN1
  layernorm_f16<<<MROWS, 256, 0, stream>>>(x, ln1g, ln1b, xn);

  // 3) QKV + bias + RoPE (z: 0=Q,1=K,2=V)
  qkv_rope_kernel<<<dim3(NH, MROWS / 64, 3), 128, 0, stream>>>(
      xn, wqT, wkT, wvT, bq, bk, bv, positions, Qb, Kb, VTb);

  // 4) masked flash attention
  attention_kernel<<<dim3(SEQ / 64, NH, BN), 128, 0, stream>>>(
      Qb, Kb, VTb, mask, attnh);

  // 5) out-projection + residual(x)  ->  x2 (f32)
  gemm_wmma<false, true, false><<<dim3(DIM / 64, MROWS / 64), 128, 0, stream>>>(
      attnh, woT, bo, x, nullptr, x2, DIM, DIM);

  // 6) LN2
  layernorm_f16<<<MROWS, 256, 0, stream>>>(x2, ln2g, ln2b, h2);

  // 7) FFN1 + ReLU (f16 out)
  gemm_wmma<true, false, true><<<dim3(FFN_DIM / 64, MROWS / 64), 128, 0, stream>>>(
      h2, w1T, b1, nullptr, ffn1, nullptr, DIM, FFN_DIM);

  // 8) FFN2 + residual(x2) -> d_out (f32)
  gemm_wmma<false, true, false><<<dim3(DIM / 64, MROWS / 64), 128, 0, stream>>>(
      ffn1, w2T, b2, x2, nullptr, outF, FFN_DIM, DIM);
}
